// ChebyNet_1_48137993453855
// MI455X (gfx1250) — matrix operand order repaired and checked by reading, not compiled, not run on hardware
//
#include <hip/hip_runtime.h>
#include <hip/hip_bf16.h>

typedef __attribute__((ext_vector_type(16))) __bf16 v16bf;
typedef __attribute__((ext_vector_type(8)))  float  v8f;

#define BM 128
#define BN 128
#define BK 32
#define PITCH 34   // bf16 elements per LDS row (padded: 68B stagger across 64 banks)

// C[N x 512-ish] = A[N x K] @ W[K x M] + bias, optional ReLU.
// Grid: (ceil(N/128), Mtiles). 256 threads = 8 wave32, each wave: 32x64 warp tile (2x4 WMMA frags).
template <bool RELU>
__global__ __launch_bounds__(256) void gemm_bf16_wmma(
    const float* __restrict__ A, int lda,
    const float* __restrict__ W, int ldw,
    const float* __restrict__ bias,
    float* __restrict__ C, int ldc,
    int N, int K)
{
    __shared__ __bf16 sA[BM * PITCH];   // [m][k]
    __shared__ __bf16 sW[BN * PITCH];   // [n][k]  (transposed at stage time)

    const int tid  = threadIdx.x;
    const int lane = tid & 31;
    const int wave = tid >> 5;      // 0..7
    const int wm   = wave & 3;      // 4 wave-rows * 32 rows
    const int wn   = wave >> 2;     // 2 wave-cols * 64 cols
    const int half = lane >> 4;     // 0: lanes 0-15, 1: lanes 16-31
    const int l16  = lane & 15;

    const int row0 = blockIdx.x * BM;
    const int col0 = blockIdx.y * BN;

    v8f acc[2][4];
    #pragma unroll
    for (int i = 0; i < 2; ++i)
        #pragma unroll
        for (int j = 0; j < 4; ++j)
            acc[i][j] = v8f{0.f,0.f,0.f,0.f,0.f,0.f,0.f,0.f};

    for (int kk = 0; kk < K; kk += BK) {
        // ---- stage A: 128 rows x 32 k (fp32 -> bf16), 1024 float4 / 256 thr = 4 each
        #pragma unroll
        for (int i = 0; i < 4; ++i) {
            int f4 = tid + i * 256;
            int r  = f4 >> 3;       // row in tile
            int c4 = f4 & 7;        // float4 index in k
            int grow = row0 + r;
            float4 v = {0.f, 0.f, 0.f, 0.f};
            if (grow < N) {
                const float* ap = A + (size_t)grow * lda + kk + c4 * 4;
                v = *(const float4*)ap;
                if (kk + BK < K) __builtin_prefetch(ap + BK, 0, 3);
            }
            __bf16* dst = &sA[r * PITCH + c4 * 4];
            dst[0] = (__bf16)v.x; dst[1] = (__bf16)v.y;
            dst[2] = (__bf16)v.z; dst[3] = (__bf16)v.w;
        }
        // ---- stage W: 32 k x 128 n, coalesced read, transposed store [n][k]
        #pragma unroll
        for (int i = 0; i < 4; ++i) {
            int f4 = tid + i * 256;
            int r  = f4 >> 5;       // k row 0..31
            int c4 = f4 & 31;       // n float4 0..31
            const float* wp = W + (size_t)(kk + r) * ldw + col0 + c4 * 4;
            float4 v = *(const float4*)wp;
            if (kk + BK < K) __builtin_prefetch(wp + (size_t)BK * ldw, 0, 3);
            sW[(c4 * 4 + 0) * PITCH + r] = (__bf16)v.x;
            sW[(c4 * 4 + 1) * PITCH + r] = (__bf16)v.y;
            sW[(c4 * 4 + 2) * PITCH + r] = (__bf16)v.z;
            sW[(c4 * 4 + 3) * PITCH + r] = (__bf16)v.w;
        }
        __syncthreads();

        // ---- A fragments: lane holds row m=l16, K = half*16 + e (contiguous, matches B)
        v16bf afrag[2];
        #pragma unroll
        for (int fm = 0; fm < 2; ++fm) {
            int m = wm * 32 + fm * 16 + l16;
            const __bf16* ap = &sA[m * PITCH];
            v16bf a;
            #pragma unroll
            for (int e = 0; e < 16; ++e) {
                int vg = e >> 1, lo = e & 1;
                int k  = half * 16 + vg * 2 + lo;
                a[e] = ap[k];
            }
            afrag[fm] = a;
        }
        // ---- B fragments: lane holds column n=l16, K = half*16 + e (contiguous in sW[n][k])
        v16bf bfrag[4];
        #pragma unroll
        for (int fn = 0; fn < 4; ++fn) {
            int n = wn * 64 + fn * 16 + l16;
            const __bf16* bp = &sW[n * PITCH + half * 16];
            v16bf b;
            #pragma unroll
            for (int e = 0; e < 16; ++e) b[e] = bp[e];
            bfrag[fn] = b;
        }

        #pragma unroll
        for (int fm = 0; fm < 2; ++fm)
            #pragma unroll
            for (int fn = 0; fn < 4; ++fn)
                acc[fm][fn] = __builtin_amdgcn_wmma_f32_16x16x32_bf16(
                    false, afrag[fm], false, bfrag[fn],
                    (short)0, acc[fm][fn], false, false);

        __syncthreads();
    }

    // ---- epilogue: C layout = lane->(N=l16), VGPR v -> M = v + 8*half
    #pragma unroll
    for (int fm = 0; fm < 2; ++fm)
        #pragma unroll
        for (int fn = 0; fn < 4; ++fn) {
            int coln = col0 + wn * 64 + fn * 16 + l16;
            float bv = bias ? bias[coln] : 0.f;
            #pragma unroll
            for (int v = 0; v < 8; ++v) {
                int rowm = row0 + wm * 32 + fm * 16 + v + half * 8;
                if (rowm < N) {
                    float r = acc[fm][fn][v] + bv;
                    if (RELU) r = fmaxf(r, 0.f);
                    C[(size_t)rowm * ldc + coln] = r;
                }
            }
        }
}

// Column sums / sums-of-squares for BN over N rows, M==512. 512 threads: one column each.
__global__ __launch_bounds__(512) void colstats(
    const float* __restrict__ X, int N,
    float* __restrict__ sum, float* __restrict__ sumsq, int rows_per_block)
{
    int c  = threadIdx.x;
    int r0 = blockIdx.x * rows_per_block;
    int r1 = min(N, r0 + rows_per_block);
    float s = 0.f, q = 0.f;
    for (int r = r0; r < r1; ++r) {
        float v = X[(size_t)r * 512 + c];
        s += v; q += v * v;
    }
    atomicAdd(&sum[c], s);
    atomicAdd(&sumsq[c], q);
}

// In-place BN (biased var, eps=1e-5) + ReLU, M==512.
__global__ __launch_bounds__(256) void bnrelu(
    float* __restrict__ X, const float* __restrict__ sum, const float* __restrict__ sumsq,
    const float* __restrict__ gamma, const float* __restrict__ beta, int N)
{
    size_t i = (size_t)blockIdx.x * blockDim.x + threadIdx.x;
    size_t total = (size_t)N * 512;
    if (i >= total) return;
    int c = (int)(i & 511);
    float inv_n = 1.f / (float)N;
    float mu  = sum[c] * inv_n;
    float var = sumsq[c] * inv_n - mu * mu;
    float y = (X[i] - mu) * rsqrtf(var + 1e-5f) * gamma[c] + beta[c];
    X[i] = fmaxf(y, 0.f);
}

// Final tiny GEMM: out[N,10] = G[N,512] @ W[512,10] + b.  W cached in LDS, one row per thread.
__global__ __launch_bounds__(256) void head10(
    const float* __restrict__ G, const float* __restrict__ W,
    const float* __restrict__ b, float* __restrict__ out, int N)
{
    __shared__ float sWt[512 * 10];
    __shared__ float sb[10];
    for (int i = threadIdx.x; i < 512 * 10; i += 256) sWt[i] = W[i];
    if (threadIdx.x < 10) sb[threadIdx.x] = b[threadIdx.x];
    __syncthreads();
    int row = blockIdx.x * 256 + threadIdx.x;
    if (row >= N) return;
    const float* g = G + (size_t)row * 512;
    float acc[10];
    #pragma unroll
    for (int j = 0; j < 10; ++j) acc[j] = sb[j];
    for (int k = 0; k < 512; k += 4) {
        float4 gv = *(const float4*)(g + k);
        #pragma unroll
        for (int j = 0; j < 10; ++j)
            acc[j] += gv.x * sWt[(k + 0) * 10 + j] + gv.y * sWt[(k + 1) * 10 + j]
                    + gv.z * sWt[(k + 2) * 10 + j] + gv.w * sWt[(k + 3) * 10 + j];
    }
    #pragma unroll
    for (int j = 0; j < 10; ++j) out[(size_t)row * 10 + j] = acc[j];
}

extern "C" void kernel_launch(void* const* d_in, const int* in_sizes, int n_in,
                              void* d_out, int out_size, void* d_ws, size_t ws_size,
                              hipStream_t stream)
{
    const int N = in_sizes[0] / 128;

    auto F = [&](int i) { return (const float*)d_in[i]; };
    const float* x1 = F(0);
    const float* x2 = F(3);

    float* buf0  = (float*)d_ws;                 // N x 512
    float* buf1  = buf0 + (size_t)N * 512;       // N x 512
    float* hcat  = buf1 + (size_t)N * 512;       // N x 1024
    float* stats = hcat + (size_t)N * 1024;      // 1024 floats (sum | sumsq)

    const dim3 blk(256);
    const int  gm = (N + BM - 1) / BM;
    const dim3 g4(gm, 4);                        // M = 512 everywhere
    const int  sb = (N + 511) / 512;             // colstats blocks
    const int  eb = (int)(((size_t)N * 512 + 255) / 256);

    // One branch: gemm1 -> BN/ReLU -> gemm2 -> BN/ReLU -> gemm3 into hcat half.
    auto branch = [&](const float* x, int base, float* outcol) {
        gemm_bf16_wmma<false><<<g4, blk, 0, stream>>>(x, 128, F(base + 0), 512, F(base + 1),
                                                      buf0, 512, N, 128);
        hipMemsetAsync(stats, 0, 1024 * sizeof(float), stream);
        colstats<<<sb, 512, 0, stream>>>(buf0, N, stats, stats + 512, 512);
        bnrelu<<<eb, blk, 0, stream>>>(buf0, stats, stats + 512, F(base + 2), F(base + 3), N);

        gemm_bf16_wmma<false><<<g4, blk, 0, stream>>>(buf0, 512, F(base + 4), 512, F(base + 5),
                                                      buf1, 512, N, 512);
        hipMemsetAsync(stats, 0, 1024 * sizeof(float), stream);
        colstats<<<sb, 512, 0, stream>>>(buf1, N, stats, stats + 512, 512);
        bnrelu<<<eb, blk, 0, stream>>>(buf1, stats, stats + 512, F(base + 6), F(base + 7), N);

        gemm_bf16_wmma<false><<<g4, blk, 0, stream>>>(buf1, 512, F(base + 8), 512, F(base + 9),
                                                      outcol, 1024, N, 512);
    };

    branch(x1, 6,  hcat);        // h1 -> hcat[:, 0:512]
    branch(x2, 16, hcat + 512);  // h2 -> hcat[:, 512:1024]

    // head: relu(hcat @ Wh1 + bh1) -> buf0 ; buf0 @ Wh2 + bh2 -> out
    gemm_bf16_wmma<true><<<g4, blk, 0, stream>>>(hcat, 1024, F(26), 512, F(27),
                                                 buf0, 512, N, 1024);
    head10<<<(N + 255) / 256, blk, 0, stream>>>(buf0, F(28), F(29), (float*)d_out, N);
}
